// VQSpeaker_30545807409756
// MI455X (gfx1250) — matrix operand order repaired and checked by reading, not compiled, not run on hardware
//
#include <hip/hip_runtime.h>
#include <hip/hip_bf16.h>
#include <cstdint>

// ---------------------------------------------------------------------------
// VQ-speaker: MLP (128->64->64->64, relu,relu,linear) + euclidean VQ argmin
// over a 2048x64 codebook.  Matmuls via v_wmma_f32_16x16x32_f16 (f32 acc);
// codebook tiles staged LDS-side with the Tensor Data Mover (double buffered).
// ---------------------------------------------------------------------------

typedef _Float16 v16h __attribute__((ext_vector_type(16)));
typedef _Float16 v8h  __attribute__((ext_vector_type(8)));
typedef float    v8f  __attribute__((ext_vector_type(8)));

union Frag { v16h v; v8h h[2]; };

static constexpr int Btot  = 32;
static constexpr int Ntot  = 2048;
static constexpr int Mrows = Btot * Ntot;   // 65536
static constexpr int INDIM = 128;
static constexpr int HID   = 64;
static constexpr int OUT   = 64;
static constexpr int CBN   = 2048;
static constexpr int TILE  = 128;           // codebook entries per LDS tile
static constexpr int NT    = CBN / TILE;    // 16 tiles

#ifndef __has_builtin
#define __has_builtin(x) 0
#endif

#if __has_builtin(__builtin_amdgcn_tensor_load_to_lds)
#define USE_TDM 1
#else
#define USE_TDM 0
#endif

__device__ __forceinline__ v8h cvt8(const float* __restrict__ p) {
  v8h r;
#pragma unroll
  for (int i = 0; i < 8; ++i) r[i] = (_Float16)p[i];
  return r;
}

__device__ __forceinline__ v8f wmma16(const Frag& a, const Frag& b, v8f c) {
  // (neg_a, A, neg_b, B, c_mod, C, reuse_a, reuse_b)
  return __builtin_amdgcn_wmma_f32_16x16x32_f16(false, a.v, false, b.v,
                                                (short)0, c, false, false);
}

#if USE_TDM
typedef unsigned int u32x4 __attribute__((ext_vector_type(4)));
typedef int          i32x4 __attribute__((ext_vector_type(4)));
typedef int          i32x8 __attribute__((ext_vector_type(8)));

// 1-D contiguous tile copy global->LDS via TDM.  nbytes % 8 == 0,
// nbytes/8 <= 65535.  Issued wave-wide; completion tracked by TENSORcnt.
__device__ __forceinline__ void tdm_load_1d(unsigned lds_off,
                                            const void* gaddr,
                                            unsigned nbytes) {
  unsigned long long ga = (unsigned long long)(uintptr_t)gaddr;
  const unsigned elems = nbytes >> 3;          // 8-byte elements
  u32x4 g0 = {};
  g0[0] = 1u;                                   // count=1, user mode, no gather
  g0[1] = lds_off;                              // D#.lds_addr (bytes)
  g0[2] = (unsigned)(ga & 0xFFFFFFFFu);         // global_addr[31:0]
  g0[3] = (unsigned)((ga >> 32) & 0x1FFFFFFu)   // global_addr[56:32]
          | (2u << 30);                         // type=2 ("image")
  i32x8 g1 = {};
  g1[0] = (3 << 16);                            // data_size=3 (8B); wg_mask=0
  const unsigned td0 = 0x40000000u;             // huge tensor_dim0: no OOB clip
  g1[1] = (int)((td0 & 0xFFFFu) << 16);         // tensor_dim0[15:0]  @bits 63:48
  g1[2] = (int)(((td0 >> 16) & 0xFFFFu)         // tensor_dim0[31:16] @bits 79:64
                | (1u << 16));                  // tensor_dim1[15:0]=1
  g1[3] = (int)(elems << 16);                   // tile_dim0 @bits 127:112
  g1[5] = (int)elems;                           // tensor_dim0_stride[31:0]
  i32x4 g2 = {};
  i32x4 g3 = {};
#if __has_include(<hip/amd_detail/amd_gfx1250_TDM.h>)
  i32x8 gx = {};
  __builtin_amdgcn_tensor_load_to_lds(g0, g1, g2, g3, gx, 0);
#else
  __builtin_amdgcn_tensor_load_to_lds(g0, g1, g2, g3, 0);
#endif
}
#endif  // USE_TDM

// ---------------------------------------------------------------------------
// Kernel 0: codebook f32 -> f16 (row major) + per-entry squared norm
// ---------------------------------------------------------------------------
__global__ __launch_bounds__(256)
void vq_prep_kernel(const float* __restrict__ cb,
                    _Float16* __restrict__ cb_h,
                    float* __restrict__ e2) {
  int c = blockIdx.x * 256 + threadIdx.x;
  if (c >= CBN) return;
  const float* src = cb + (size_t)c * OUT;
  float s = 0.f;
#pragma unroll
  for (int j = 0; j < OUT; ++j) {
    float v = src[j];
    s += v * v;
    cb_h[(size_t)c * OUT + j] = (_Float16)v;
  }
  e2[c] = s;
}

// ---------------------------------------------------------------------------
// Kernel 1: fused 3-layer MLP.  8 waves/block, 16 rows/wave, 128 rows/block.
// Weights live in LDS as f16, column-major (K contiguous per output column)
// so B-matrix fragments are two contiguous 16B per-lane loads.
// ---------------------------------------------------------------------------
__global__ __launch_bounds__(256)
void mlp_kernel(const float* __restrict__ obs,
                const float* __restrict__ W1, const float* __restrict__ b1,
                const float* __restrict__ W2, const float* __restrict__ b2,
                const float* __restrict__ W3, const float* __restrict__ b3,
                float* __restrict__ x_f32, _Float16* __restrict__ x_f16) {
  __shared__ __align__(32) _Float16 sW1[HID * INDIM];  // [n][k] 16KB
  __shared__ __align__(32) _Float16 sW2[HID * HID];    // 8KB
  __shared__ __align__(32) _Float16 sW3[HID * OUT];    // 8KB
  __shared__ float sb1[HID], sb2[HID], sb3[OUT];
  __shared__ __align__(32) _Float16 sH[8][16 * HID];   // per-wave activ. 16KB

  const int tid = threadIdx.x;
  for (int i = tid; i < INDIM * HID; i += 256) {
    int k = i >> 6, n = i & 63;                  // W1 row-major [k][n]
    sW1[n * INDIM + k] = (_Float16)W1[i];
  }
  for (int i = tid; i < HID * HID; i += 256) {
    int k = i >> 6, n = i & 63;
    sW2[n * HID + k] = (_Float16)W2[i];
    sW3[n * HID + k] = (_Float16)W3[i];
  }
  if (tid < 64) { sb1[tid] = b1[tid]; sb2[tid] = b2[tid]; sb3[tid] = b3[tid]; }
  __syncthreads();

  const int wave = tid >> 5, lane = tid & 31;
  const int half = lane >> 4, l16 = lane & 15;
  const int row0 = blockIdx.x * 128 + wave * 16;
  _Float16* hb = &sH[wave][0];

  // ---- layer 1: A fragments straight from obs (f32 -> f16 convert) ----
  // ISA A layout (16x32 f16): lanes 0-15 row=l16 K{0..7,16..23},
  //                           lanes 16-31 row=l16 K{8..15,24..31}
  Frag a1[4];
  const float* orow = obs + (size_t)(row0 + l16) * INDIM;
#pragma unroll
  for (int kc = 0; kc < 4; ++kc) {
    int koff = kc * 32 + half * 8;
    a1[kc].h[0] = cvt8(orow + koff);
    a1[kc].h[1] = cvt8(orow + koff + 16);
  }
#pragma unroll
  for (int nt = 0; nt < 4; ++nt) {
    v8f acc = {};
    const int col = nt * 16 + l16;
#pragma unroll
    for (int kc = 0; kc < 4; ++kc) {
      Frag bfr;
      const _Float16* bp = &sW1[col * INDIM + kc * 32 + half * 8];
      bfr.h[0] = *(const v8h*)bp;
      bfr.h[1] = *(const v8h*)(bp + 16);
      acc = wmma16(a1[kc], bfr, acc);
    }
    float bias = sb1[col];
#pragma unroll
    for (int e = 0; e < 8; ++e) {   // C layout: row = e + 8*half, col = col
      float v = acc[e] + bias;
      v = v > 0.f ? v : 0.f;
      hb[(e + half * 8) * HID + col] = (_Float16)v;
    }
  }

  // ---- layer 2 ----
  Frag a2[2];
#pragma unroll
  for (int kc = 0; kc < 2; ++kc) {
    const _Float16* hp = hb + l16 * HID + kc * 32 + half * 8;
    a2[kc].h[0] = *(const v8h*)hp;
    a2[kc].h[1] = *(const v8h*)(hp + 16);
  }
#pragma unroll
  for (int nt = 0; nt < 4; ++nt) {
    v8f acc = {};
    const int col = nt * 16 + l16;
#pragma unroll
    for (int kc = 0; kc < 2; ++kc) {
      Frag bfr;
      const _Float16* bp = &sW2[col * HID + kc * 32 + half * 8];
      bfr.h[0] = *(const v8h*)bp;
      bfr.h[1] = *(const v8h*)(bp + 16);
      acc = wmma16(a2[kc], bfr, acc);
    }
    float bias = sb2[col];
#pragma unroll
    for (int e = 0; e < 8; ++e) {
      float v = acc[e] + bias;
      v = v > 0.f ? v : 0.f;
      hb[(e + half * 8) * HID + col] = (_Float16)v;  // safe: a2 already read
    }
  }

  // ---- layer 3 (no relu) -> x, spilled as f32 and f16 ----
  Frag a3[2];
#pragma unroll
  for (int kc = 0; kc < 2; ++kc) {
    const _Float16* hp = hb + l16 * HID + kc * 32 + half * 8;
    a3[kc].h[0] = *(const v8h*)hp;
    a3[kc].h[1] = *(const v8h*)(hp + 16);
  }
#pragma unroll
  for (int nt = 0; nt < 4; ++nt) {
    v8f acc = {};
    const int col = nt * 16 + l16;
#pragma unroll
    for (int kc = 0; kc < 2; ++kc) {
      Frag bfr;
      const _Float16* bp = &sW3[col * HID + kc * 32 + half * 8];
      bfr.h[0] = *(const v8h*)bp;
      bfr.h[1] = *(const v8h*)(bp + 16);
      acc = wmma16(a3[kc], bfr, acc);
    }
    float bias = sb3[col];
#pragma unroll
    for (int e = 0; e < 8; ++e) {
      float v = acc[e] + bias;
      size_t row = (size_t)(row0 + e + half * 8);
      x_f32[row * OUT + col] = v;
      x_f16[row * OUT + col] = (_Float16)v;
    }
  }
}

// ---------------------------------------------------------------------------
// Kernel 2: VQ.  argmin_c (|e_c|^2 - 2 x.e_c)  (|x|^2 is row-constant).
// Codebook staged into LDS in 128-entry tiles, DOUBLE BUFFERED via the
// Tensor Data Mover (wave 0 issues, TENSORcnt + barrier publishes) so the
// copy of tile T+1 overlaps WMMA compute on tile T.  Per-lane running
// (best,idx); 16-lane shuffle min-reduce; f32 gather for msg; deterministic
// per-block loss partial.
// ---------------------------------------------------------------------------
__global__ __launch_bounds__(256)
void vq_kernel(const float* __restrict__ codebook,
               const _Float16* __restrict__ cb_h_g,
               const float* __restrict__ e2_g,
               const float* __restrict__ x_f32,
               const _Float16* __restrict__ x_f16,
               float* __restrict__ out_msg,
               int* __restrict__ out_idx,
               float* __restrict__ block_loss) {
  __shared__ __align__(32) _Float16 sCB[2][TILE * OUT];  // 2 x 16KB tiles
  __shared__ float sE2[2][TILE];
  __shared__ int   sIdx[8][16];
  __shared__ float sRed[256];

  const int tid = threadIdx.x, wave = tid >> 5, lane = tid & 31;
  const int half = lane >> 4, l16 = lane & 15;
  const int row0 = blockIdx.x * 128 + wave * 16;

  // A fragments of x (f16), resident for the whole codebook sweep
  Frag xa[2];
  const _Float16* xrow = x_f16 + (size_t)(row0 + l16) * OUT;
#pragma unroll
  for (int kc = 0; kc < 2; ++kc) {
    int koff = kc * 32 + half * 8;
    xa[kc].h[0] = *(const v8h*)(xrow + koff);
    xa[kc].h[1] = *(const v8h*)(xrow + koff + 16);
  }

  float best[8];
  int   bidx[8];
#pragma unroll
  for (int e = 0; e < 8; ++e) { best[e] = 3.4e38f; bidx[e] = 0; }

  constexpr unsigned TILE_BYTES = TILE * OUT * sizeof(_Float16);  // 16384

#if USE_TDM
  // prologue: kick off tile 0 into buffer 0 (wave 0 only; EXEC-independent DMA)
  if (tid < 32) {
    tdm_load_1d((unsigned)(uintptr_t)(void*)&sCB[0][0], cb_h_g, TILE_BYTES);
  }
#endif

  for (int T = 0; T < NT; ++T) {
    const int cur = T & 1;
    __syncthreads();  // readers of buf[1-cur] (tile T-1) are done -> reusable
#if USE_TDM
    if (tid < 32) {
      if (T + 1 < NT) {
        tdm_load_1d((unsigned)(uintptr_t)(void*)&sCB[1 - cur][0],
                    cb_h_g + (size_t)(T + 1) * TILE * OUT, TILE_BYTES);
        __builtin_amdgcn_s_wait_tensorcnt(1);  // tile T arrived (in-order)
      } else {
        __builtin_amdgcn_s_wait_tensorcnt(0);
      }
    }
#else
    {
      const uint4* src = (const uint4*)(cb_h_g + (size_t)T * TILE * OUT);
      uint4* dst = (uint4*)&sCB[cur][0];
      for (int i = tid; i < (int)(TILE_BYTES / 16); i += 256) dst[i] = src[i];
    }
#endif
    if (tid < TILE) sE2[cur][tid] = e2_g[T * TILE + tid];
    __syncthreads();  // tile T + e2 visible to all waves

#pragma unroll 4
    for (int ct = 0; ct < TILE / 16; ++ct) {
      v8f acc = {};
      const int ent = ct * 16 + l16;  // B col = lane&15
#pragma unroll
      for (int kc = 0; kc < 2; ++kc) {
        Frag bfr;
        const _Float16* bp = &sCB[cur][ent * OUT + kc * 32 + half * 8];
        bfr.h[0] = *(const v8h*)bp;
        bfr.h[1] = *(const v8h*)(bp + 16);
        acc = wmma16(xa[kc], bfr, acc);
      }
      const float e2v = sE2[cur][ent];
      const int cglob = T * TILE + ent;
#pragma unroll
      for (int e = 0; e < 8; ++e) {
        float s = e2v - 2.f * acc[e];
        if (s < best[e]) { best[e] = s; bidx[e] = cglob; }  // strict<: first min
      }
    }
  }

  // min-reduce across the 16 lanes sharing each row (xor stays within half)
#pragma unroll
  for (int e = 0; e < 8; ++e) {
    float v = best[e];
    int id = bidx[e];
#pragma unroll
    for (int off = 1; off < 16; off <<= 1) {
      float ov = __shfl_xor(v, off, 32);
      int  oid = __shfl_xor(id, off, 32);
      if (ov < v || (ov == v && oid < id)) { v = ov; id = oid; }
    }
    if (l16 == 0) {
      int rl = e + half * 8;
      sIdx[wave][rl] = id;
      out_idx[row0 + rl] = id;
    }
  }
  __syncthreads();

  // msg = codebook[idx] (exact f32 gather); loss partial vs f32 x
  float lsum = 0.f;
  for (int r = 0; r < 16; ++r) {
    const int id = sIdx[wave][r];
    const size_t row = (size_t)(row0 + r);
#pragma unroll
    for (int j = lane; j < OUT; j += 32) {
      float q = codebook[(size_t)id * OUT + j];
      float xv = x_f32[row * OUT + j];
      float d = q - xv;
      lsum += d * d;
      out_msg[row * OUT + j] = q;
    }
  }
  sRed[tid] = lsum;
  __syncthreads();
  for (int s = 128; s > 0; s >>= 1) {
    if (tid < s) sRed[tid] += sRed[tid + s];
    __syncthreads();
  }
  if (tid == 0) block_loss[blockIdx.x] = sRed[0];
}

// ---------------------------------------------------------------------------
// Kernel 3: deterministic final loss reduction (512 partials -> mean)
// ---------------------------------------------------------------------------
__global__ __launch_bounds__(256)
void loss_kernel(const float* __restrict__ block_loss, float* __restrict__ out_loss) {
  __shared__ float red[256];
  red[threadIdx.x] = block_loss[threadIdx.x] + block_loss[threadIdx.x + 256];
  __syncthreads();
  for (int s = 128; s > 0; s >>= 1) {
    if (threadIdx.x < s) red[threadIdx.x] += red[threadIdx.x + s];
    __syncthreads();
  }
  if (threadIdx.x == 0)
    *out_loss = red[0] * (1.0f / ((float)Mrows * (float)OUT));
}

// ---------------------------------------------------------------------------
extern "C" void kernel_launch(void* const* d_in, const int* in_sizes, int n_in,
                              void* d_out, int out_size, void* d_ws, size_t ws_size,
                              hipStream_t stream) {
  const float* obs = (const float*)d_in[0];
  const float* W1  = (const float*)d_in[1];
  const float* b1  = (const float*)d_in[2];
  const float* W2  = (const float*)d_in[3];
  const float* b2  = (const float*)d_in[4];
  const float* W3  = (const float*)d_in[5];
  const float* b3  = (const float*)d_in[6];
  const float* cb  = (const float*)d_in[7];

  char* ws = (char*)d_ws;
  _Float16* x_h   = (_Float16*)(ws);                       //  8,388,608 B
  float*    x_f   = (float*)(ws + 8388608);                // 16,777,216 B
  _Float16* cb_h  = (_Float16*)(ws + 25165824);            //    262,144 B
  float*    e2    = (float*)(ws + 25165824 + 262144);      //      8,192 B
  float*    bloss = (float*)(ws + 25165824 + 262144 + 8192);

  // tuple output flat-concat: msg[65536*64] f32 | idx[65536] i32 | loss[1] f32
  float* out_msg  = (float*)d_out;
  int*   out_idx  = (int*)((float*)d_out + (size_t)Mrows * OUT);
  float* out_loss = (float*)d_out + (size_t)Mrows * OUT + Mrows;

  vq_prep_kernel<<<CBN / 256, 256, 0, stream>>>(cb, cb_h, e2);
  mlp_kernel<<<Mrows / 128, 256, 0, stream>>>(obs, W1, b1, W2, b2, W3, b3, x_f, x_h);
  vq_kernel<<<Mrows / 128, 256, 0, stream>>>(cb, cb_h, e2, x_f, x_h,
                                             out_msg, out_idx, bloss);
  loss_kernel<<<1, 256, 0, stream>>>(bloss, out_loss);
}
